// TransformerAttention_69664369541761
// MI455X (gfx1250) — compile-verified, matrix operand order checked by reading, not run on hardware
//
#include <hip/hip_runtime.h>

// ---------------- problem constants ----------------
#define B_BATCH 4
#define L_SEQ   2048
#define H_HID   512
#define N_HEADS 8
#define D_HEAD  64
#define RP      (2 * L_SEQ - 1)   // 4095
#define M_ROWS  (B_BATCH * L_SEQ) // 8192

typedef __attribute__((ext_vector_type(16))) __bf16 v16bf;
typedef __attribute__((ext_vector_type(8)))  float  v8f;
typedef __attribute__((ext_vector_type(8)))  unsigned short u16x8;

union BF16Frag { v16bf v; unsigned short u[16]; };
union AccFrag  { v8f   v; float f[8]; };

#ifndef __has_builtin
#define __has_builtin(x) 0
#endif
#if __has_builtin(__builtin_amdgcn_global_load_async_to_lds_b128) && \
    __has_builtin(__builtin_amdgcn_s_wait_asynccnt)
#define HAVE_ASYNC_LDS 1
#else
#define HAVE_ASYNC_LDS 0
#endif

#if HAVE_ASYNC_LDS
// builtin parameter type: 16-byte int vector pointer in AS1 (src) / AS3 (dst)
typedef int async_b128_t __attribute__((vector_size(16)));
typedef __attribute__((address_space(1))) async_b128_t* async_gptr;
typedef __attribute__((address_space(3))) async_b128_t* async_lptr;
#endif

// round-to-nearest-even f32 -> bf16 (bit pattern)
__device__ __forceinline__ unsigned short f32_to_bf16(float f) {
  unsigned int u = __float_as_uint(f);
  unsigned int r = u + 0x7fffu + ((u >> 16) & 1u);
  return (unsigned short)(r >> 16);
}

// K-index inside a 16-bit WMMA A/B fragment (ISA 7.12.2: 16-bit A 16x32 layout)
__device__ __forceinline__ int frag_k(int e, int hi) {
  int v = e >> 1, p = e & 1;
  return (v < 4 ? 2 * v : 16 + 2 * (v - 4)) + p + (hi ? 8 : 0);
}

// Build a fragment from a K-contiguous (row-major) bf16 row: p = row + chunk*32.
// Elements are two contiguous 8-half runs -> two 16B vector loads.
__device__ __forceinline__ BF16Frag load_frag_krow(const unsigned short* p, int hi) {
  BF16Frag f;
  u16x8 lo = *(const u16x8*)(p + hi * 8);
  u16x8 hi8 = *(const u16x8*)(p + 16 + hi * 8);
#pragma unroll
  for (int i = 0; i < 8; ++i) { f.u[i] = lo[i]; f.u[8 + i] = hi8[i]; }
  return f;
}

// ---------------------------------------------------------------------------
// Pack a row-major f32 matrix [R x C] (C % 32 == 0) into WMMA-fragment-packed
// bf16: dst[((rt*KC + kc)*32 + lane)*16 + e] = bf16(src[(rt*16 + lane%16)*C +
// kc*32 + frag_k(e, lane/16)]).  Works for A operands (rows = M) and for
// weights W used as B = W^T (rows = N) — same register layout.
// ---------------------------------------------------------------------------
__global__ void pack_mat_frag_kernel(const float* __restrict__ src,
                                     unsigned short* __restrict__ dst, int C) {
  unsigned idx = blockIdx.x * 256 + threadIdx.x;   // over R*C
  unsigned e = idx & 15u, lane = (idx >> 4) & 31u;
  unsigned KC = (unsigned)C >> 5;
  unsigned kc = (idx >> 9) % KC;
  unsigned rt = (idx >> 9) / KC;
  unsigned row = rt * 16 + (lane & 15u);
  unsigned col = kc * 32 + frag_k(e, lane >> 4);
  dst[idx] = f32_to_bf16(src[(size_t)row * C + col]);
}

// ---------------------------------------------------------------------------
// Head-split fragment pack: src (B,L,H) f32 -> dst fragment-packed bf16 over
// rows (bh*L + i), cols d in [0,64). Same layout serves Q (A operand, M=i)
// and K^T (B operand, N=j).
// ---------------------------------------------------------------------------
__global__ void pack_head_frag_kernel(const float* __restrict__ src,
                                      unsigned short* __restrict__ dst) {
  unsigned idx = blockIdx.x * 256 + threadIdx.x;   // over B*H*L*D
  unsigned e = idx & 15u, lane = (idx >> 4) & 31u, c = (idx >> 9) & 1u;
  unsigned rt = idx >> 10;
  unsigned iblk = rt & 127u, bh = rt >> 7;
  unsigned b = bh >> 3, h = bh & 7u;
  unsigned row = iblk * 16 + (lane & 15u);
  unsigned d = c * 32 + frag_k(e, lane >> 4);
  dst[idx] = f32_to_bf16(src[((size_t)b * L_SEQ + row) * H_HID + h * D_HEAD + d]);
}

__global__ void pack_cast_kernel(const float* __restrict__ src,
                                 unsigned short* __restrict__ dst, int n) {
  int i = blockIdx.x * 256 + threadIdx.x;
  if (i < n) dst[i] = f32_to_bf16(src[i]);
}

// ---------------------------------------------------------------------------
// GEMM on fragment-packed operands: C[M,N] = scale*(A @ W^T + bias).
// block = 8 waves; wave -> 16 rows x 64 cols. Inner loop: 2+8 b128 loads +
// 4 WMMAs per 32-wide K chunk, zero conversion VALU.
// ---------------------------------------------------------------------------
__global__ __launch_bounds__(256) void gemm_packed_wmma(
    const unsigned short* __restrict__ Ap, const unsigned short* __restrict__ Wp,
    const float* __restrict__ bias, float* __restrict__ C_out,
    int N, int K, float scale)
{
  const int wave = threadIdx.x >> 5;
  const int lane = threadIdx.x & 31;
  const int hi = lane >> 4, lm = lane & 15;
  const int mt = blockIdx.y * 8 + wave;
  const int nt0 = blockIdx.x * 4;
  const int KC = K >> 5;

  AccFrag acc[4] = {};
  const unsigned short* ap = Ap + ((size_t)mt * KC * 32 + lane) * 16;

  for (int kc = 0; kc < KC; ++kc) {
    BF16Frag a;
    a.v = *(const v16bf*)(ap + (size_t)kc * 512);
    if (kc + 1 < KC) __builtin_prefetch(ap + (size_t)(kc + 1) * 512, 0, 3);
#pragma unroll
    for (int t = 0; t < 4; ++t) {
      BF16Frag b;
      b.v = *(const v16bf*)(Wp + (((size_t)(nt0 + t) * KC + kc) * 32 + lane) * 16);
      acc[t].v = __builtin_amdgcn_wmma_f32_16x16x32_bf16(
          false, a.v, false, b.v, (short)0, acc[t].v, false, false);
    }
  }

#pragma unroll
  for (int t = 0; t < 4; ++t) {
    const int n = (nt0 + t) * 16 + lm;
    const float bv = bias[n];
#pragma unroll
    for (int r = 0; r < 8; ++r) {
      const int row = mt * 16 + r + 8 * hi;
      C_out[(size_t)row * N + n] = scale * (acc[t].f[r] + bv);
    }
  }
}

// ---------------------------------------------------------------------------
// diag_softmax: one wave per 16-row block of one (b,h).
//   scores[i,j] = q_i.k_j + q_i.Wp[j-i+L-1] + bp[j-i+L-1]
//   diag[b,h,i] = exp(s[i,i]-m_i)/sum_j exp(s[i,j]-m_i)    (online softmax)
// K fragments stream through LDS via async-to-LDS double buffering.
// ---------------------------------------------------------------------------
__global__ __launch_bounds__(32) void diag_softmax_kernel(
    const unsigned short* __restrict__ qp,   // fragment-packed Q (scaled)
    const unsigned short* __restrict__ ktp,  // fragment-packed K^T
    const unsigned short* __restrict__ wpb,  // (RP, 64) bf16 row-major
    const float* __restrict__ wp_b,          // (RP) f32
    float* __restrict__ diagv)               // (B,H,L) f32
{
  const int blk  = blockIdx.x;               // bh * (L/16) + iblk
  const int iblk = blk & 127;                // L/16 = 128
  const int bh   = blk >> 7;
  const int i0   = iblk * 16;
  const int lane = threadIdx.x;
  const int hi   = lane >> 4;
  const int lm   = lane & 15;

  __shared__ float lds_pv[16 * 32];
#if HAVE_ASYNC_LDS
  __shared__ unsigned short kstage[2][1024];  // 2 x 2KB double buffer
#endif

  // A fragments (16 q rows, K=64 in two chunks): contiguous 32B loads
  const int rt = bh * 128 + iblk;
  BF16Frag a0, a1;
  a0.v = *(const v16bf*)(qp + (size_t)rt * 1024 + lane * 16);
  a1.v = *(const v16bf*)(qp + (size_t)rt * 1024 + 512 + lane * 16);

  float m[8], ssum[8];
#pragma unroll
  for (int r = 0; r < 8; ++r) { m[r] = -3.0e38f; ssum[r] = 0.0f; }
  float diag_s = 0.0f;
  const int my_diag_r = (hi == 0) ? (lm < 8 ? lm : -1) : (lm >= 8 ? lm - 8 : -1);

  const unsigned short* kbh = ktp + (size_t)bh * 128 * 1024;  // this head's tiles

#if HAVE_ASYNC_LDS
  {
    const char* g = (const char*)kbh + lane * 16;
    char* l = (char*)&kstage[0][0] + lane * 16;
#pragma unroll
    for (int t = 0; t < 4; ++t)
      __builtin_amdgcn_global_load_async_to_lds_b128(
          (async_gptr)(g + t * 512), (async_lptr)(l + t * 512), 0, 0);
  }
#endif

  for (int jt = 0; jt < 128; ++jt) {
    const int j0 = jt * 16;

#if HAVE_ASYNC_LDS
    if (jt + 1 < 128) {  // prefetch next K tile, then wait for current
      const char* g = (const char*)kbh + (size_t)(jt + 1) * 2048 + lane * 16;
      char* l = (char*)&kstage[(jt + 1) & 1][0] + lane * 16;
#pragma unroll
      for (int t = 0; t < 4; ++t)
        __builtin_amdgcn_global_load_async_to_lds_b128(
            (async_gptr)(g + t * 512), (async_lptr)(l + t * 512), 0, 0);
      __builtin_amdgcn_s_wait_asynccnt(4);
    } else {
      __builtin_amdgcn_s_wait_asynccnt(0);
    }
    const unsigned short* kt = &kstage[jt & 1][0];
#else
    const unsigned short* kt = kbh + (size_t)jt * 1024;
#endif

    // ---- qk tile: Q(16x64) @ K^T(64x16) -> 2 WMMAs
    AccFrag c = {};
    {
      BF16Frag b0, b1;
      b0.v = *(const v16bf*)(kt + lane * 16);
      b1.v = *(const v16bf*)(kt + 512 + lane * 16);
      c.v = __builtin_amdgcn_wmma_f32_16x16x32_bf16(false, a0.v, false, b0.v,
                                                    (short)0, c.v, false, false);
      c.v = __builtin_amdgcn_wmma_f32_16x16x32_bf16(false, a1.v, false, b1.v,
                                                    (short)0, c.v, false, false);
    }

    // ---- positional window: pvec = Q(16x64) @ Wp^T[:, rbase..rbase+31]
    const int rbase = j0 - i0 - 15 + (L_SEQ - 1);
    AccFrag pv[2];
#pragma unroll
    for (int nt = 0; nt < 2; ++nt) {
      int col = rbase + nt * 16 + lm;
      col = col < 0 ? 0 : (col > RP - 1 ? RP - 1 : col);  // pad cols unused
      const unsigned short* prow = wpb + (size_t)col * 64;
      BF16Frag w0 = load_frag_krow(prow, hi);
      BF16Frag w1 = load_frag_krow(prow + 32, hi);
      AccFrag t = {};
      t.v = __builtin_amdgcn_wmma_f32_16x16x32_bf16(false, a0.v, false, w0.v,
                                                    (short)0, t.v, false, false);
      t.v = __builtin_amdgcn_wmma_f32_16x16x32_bf16(false, a1.v, false, w1.v,
                                                    (short)0, t.v, false, false);
      pv[nt] = t;
    }

    // ---- scatter pvec (C layout) to LDS for the skew gather
    __syncthreads();
#pragma unroll
    for (int r = 0; r < 8; ++r) {
      int row = r + 8 * hi;
      lds_pv[row * 32 + lm]      = pv[0].f[r];
      lds_pv[row * 32 + 16 + lm] = pv[1].f[r];
    }
    __syncthreads();

    // ---- assemble scores + online softmax (row = 16-lane group per VGPR)
    const int n = lm;
#pragma unroll
    for (int r = 0; r < 8; ++r) {
      int row = r + 8 * hi;
      int rel = (j0 + n) - (i0 + row) + (L_SEQ - 1);     // in [0, RP-1]
      float pos = lds_pv[row * 32 + (n - row + 15)] + wp_b[rel];
      float s = c.f[r] + pos;
      if (j0 == i0 && my_diag_r == r && n == row) diag_s = s;

      float tmax = s;
#pragma unroll
      for (int off = 1; off < 16; off <<= 1)
        tmax = fmaxf(tmax, __shfl_xor(tmax, off, 16));
      float mnew = fmaxf(m[r], tmax);
      ssum[r] = ssum[r] * __expf(m[r] - mnew) + __expf(s - mnew);
      m[r] = mnew;
    }
  }

  // final row sums (distributed partials -> 16-lane butterfly add)
#pragma unroll
  for (int r = 0; r < 8; ++r) {
    float t = ssum[r];
#pragma unroll
    for (int off = 1; off < 16; off <<= 1) t += __shfl_xor(t, off, 16);
    ssum[r] = t;
  }

  if (my_diag_r >= 0) {
    int row = my_diag_r + 8 * hi;
    float d = __expf(diag_s - m[my_diag_r]) / ssum[my_diag_r];
    diagv[(size_t)bh * L_SEQ + i0 + row] = d;
  }
}

// ---------------------------------------------------------------------------
// ctx[b,i,h*64+d] = diag[b,h,i] * v[b,i,h*64+d]
// ---------------------------------------------------------------------------
__global__ void scale_v_kernel(const float* __restrict__ diagv,
                               const float* __restrict__ vf,
                               float* __restrict__ ctx) {
  unsigned idx = blockIdx.x * 256 + threadIdx.x;       // over B*L*H
  unsigned cidx = idx & 511u, i = (idx >> 9) & 2047u, b = idx >> 20;
  unsigned h = cidx >> 6;
  ctx[idx] = diagv[((size_t)b * N_HEADS + h) * L_SEQ + i] * vf[idx];
}

// ---------------- workspace layout (bytes) ----------------
#define SZ_F32  ((size_t)M_ROWS * H_HID * 4)   // 16 MiB
#define SZ_BF16 ((size_t)M_ROWS * H_HID * 2)   // 8 MiB
static const size_t OFF_QF   = 0;                        // q f32 (reused as ctx)
static const size_t OFF_KF   = SZ_F32;
static const size_t OFF_VF   = 2 * SZ_F32;
static const size_t OFF_AP   = 3 * SZ_F32;               // packed activation (reused)
static const size_t OFF_WP   = OFF_AP + SZ_BF16;         // packed weight (reused)
static const size_t OFF_QP   = OFF_WP + 524288;          // packed Q frags
static const size_t OFF_KTP  = OFF_QP + SZ_BF16;         // packed K^T frags
static const size_t OFF_WPB  = OFF_KTP + SZ_BF16;        // Wp bf16 row-major
static const size_t OFF_DIAG = OFF_WPB + 524288;         // diag f32 (B,H,L)

extern "C" void kernel_launch(void* const* d_in, const int* in_sizes, int n_in,
                              void* d_out, int out_size, void* d_ws, size_t ws_size,
                              hipStream_t stream) {
  const float* q_in = (const float*)d_in[0];
  const float* k_in = (const float*)d_in[1];
  const float* v_in = (const float*)d_in[2];
  const float* Wq_w = (const float*)d_in[3];
  const float* Wq_b = (const float*)d_in[4];
  const float* Wk_w = (const float*)d_in[5];
  const float* Wk_b = (const float*)d_in[6];
  const float* Wv_w = (const float*)d_in[7];
  const float* Wv_b = (const float*)d_in[8];
  const float* Wo_w = (const float*)d_in[9];
  const float* Wo_b = (const float*)d_in[10];
  const float* Wp_w = (const float*)d_in[11];
  const float* Wp_b = (const float*)d_in[12];

  char* ws = (char*)d_ws;
  float*          qf  = (float*)(ws + OFF_QF);
  float*          kf  = (float*)(ws + OFF_KF);
  float*          vf  = (float*)(ws + OFF_VF);
  unsigned short* AP  = (unsigned short*)(ws + OFF_AP);
  unsigned short* WP  = (unsigned short*)(ws + OFF_WP);
  unsigned short* QP  = (unsigned short*)(ws + OFF_QP);
  unsigned short* KTP = (unsigned short*)(ws + OFF_KTP);
  unsigned short* WPB = (unsigned short*)(ws + OFF_WPB);
  float*          dgv = (float*)(ws + OFF_DIAG);
  float*          ctx = qf;                     // reuse qf after QP is packed
  float*          out = (float*)d_out;

  const dim3 ggemm(H_HID / 64, M_ROWS / 128);   // (8, 64)
  const int  actBlocks  = (M_ROWS * H_HID) / 256;   // 16384
  const int  wBlocks    = (H_HID * H_HID) / 256;    // 1024
  const int  headBlocks = (B_BATCH * N_HEADS * L_SEQ * D_HEAD) / 256;
  const int  wpbTotal   = RP * D_HEAD;

  // ---- Q projection (1/sqrt(D) folded in: scale applies after bias, as ref)
  pack_mat_frag_kernel<<<actBlocks, 256, 0, stream>>>(q_in, AP, H_HID);
  pack_mat_frag_kernel<<<wBlocks, 256, 0, stream>>>(Wq_w, WP, H_HID);
  gemm_packed_wmma<<<ggemm, 256, 0, stream>>>(AP, WP, Wq_b, qf, H_HID, H_HID, 0.125f);
  // ---- K projection
  pack_mat_frag_kernel<<<actBlocks, 256, 0, stream>>>(k_in, AP, H_HID);
  pack_mat_frag_kernel<<<wBlocks, 256, 0, stream>>>(Wk_w, WP, H_HID);
  gemm_packed_wmma<<<ggemm, 256, 0, stream>>>(AP, WP, Wk_b, kf, H_HID, H_HID, 1.0f);
  // ---- V projection
  pack_mat_frag_kernel<<<actBlocks, 256, 0, stream>>>(v_in, AP, H_HID);
  pack_mat_frag_kernel<<<wBlocks, 256, 0, stream>>>(Wv_w, WP, H_HID);
  gemm_packed_wmma<<<ggemm, 256, 0, stream>>>(AP, WP, Wv_b, vf, H_HID, H_HID, 1.0f);

  // ---- fragment repacks for the flash pass
  pack_head_frag_kernel<<<headBlocks, 256, 0, stream>>>(qf, QP);
  pack_head_frag_kernel<<<headBlocks, 256, 0, stream>>>(kf, KTP);
  pack_cast_kernel<<<(wpbTotal + 255) / 256, 256, 0, stream>>>(Wp_w, WPB, wpbTotal);

  // ---- streaming diagonal-softmax (never materializes LxL or pos)
  diag_softmax_kernel<<<B_BATCH * N_HEADS * (L_SEQ / 16), 32, 0, stream>>>(
      QP, KTP, WPB, Wp_b, dgv);

  // ---- out = (diag * v) @ Wo^T + Wo_b
  scale_v_kernel<<<actBlocks, 256, 0, stream>>>(dgv, vf, ctx);
  pack_mat_frag_kernel<<<actBlocks, 256, 0, stream>>>(ctx, AP, H_HID);
  pack_mat_frag_kernel<<<wBlocks, 256, 0, stream>>>(Wo_w, WP, H_HID);
  gemm_packed_wmma<<<ggemm, 256, 0, stream>>>(AP, WP, Wo_b, out, H_HID, H_HID, 1.0f);
}
